// DGCNNEncoder_38611755991128
// MI455X (gfx1250) — matrix operand order repaired and checked by reading, not compiled, not run on hardware
//
#include <hip/hip_runtime.h>
#include <hip/hip_bf16.h>
#include <math.h>

// ---------------------------------------------------------------------------
// DGCNN encoder for MI455X (gfx1250, wave32).
// GEMMs via v_wmma_f32_16x16x32_f16 (16xNd strip per wave, swizzled LDS B),
// scatter via no-return global_atomic_add_f32 (L2-resident tables).
// ---------------------------------------------------------------------------

typedef __attribute__((ext_vector_type(16))) _Float16 v16h;
typedef __attribute__((ext_vector_type(8)))  _Float16 v8h;
typedef __attribute__((ext_vector_type(8)))  float    v8f;

#define TOTAL_NODES 65536
#define GRAPHS      32
#define NODES_PER_G 2048
#define KSORT       16
#define DMAX        128

// ---------------- utility kernels ----------------

__global__ void k_zero_f32(float* p, int n) {
    int i = blockIdx.x * blockDim.x + threadIdx.x;
    if (i < n) p[i] = 0.0f;
}

__global__ void k_degrees(const int* __restrict__ src, const int* __restrict__ dst,
                          float* __restrict__ outd, float* __restrict__ ind, int E) {
    int e = blockIdx.x * blockDim.x + threadIdx.x;
    if (e < E) {
        atomicAdd(&outd[src[e]], 1.0f);
        atomicAdd(&ind[dst[e]], 1.0f);
    }
}

__global__ void k_rsqrt_deg(float* __restrict__ d, int n) {
    int i = blockIdx.x * blockDim.x + threadIdx.x;
    if (i < n) {
        float v = d[i];
        d[i] = (v > 0.0f) ? rsqrtf(fmaxf(v, 1.0f)) : 0.0f;
    }
}

__global__ void k_f32_to_f16(const float* __restrict__ in, _Float16* __restrict__ out, int n) {
    int i = blockIdx.x * blockDim.x + threadIdx.x;
    if (i < n) out[i] = (_Float16)in[i];
}

// ---------------- WMMA GEMM:  P[m,n] = od[m] * sum_k X[m,k]*W[k,n] ----------------
// X: [M x 128] f16 row-major. Wg: [128 x Nd] f16 row-major. P: [M x Nd] f32.
// Each wave computes a 16 x Nd strip (NT 16-wide tiles), reusing its A fragment
// across all N-tiles. 8 waves/block -> 8 consecutive M-tiles per block.
// The full W is staged in LDS, pre-swizzled so that each lane's B fragment
// (column n=lane&15, K = kb*32 + 16g + t, t=0..15) is 32 contiguous bytes
// -> two ds_load_b128 per fragment.
template<int NT>
__global__ __launch_bounds__(256) void k_gemm_wmma(
    const _Float16* __restrict__ X, const _Float16* __restrict__ Wg,
    const float* __restrict__ od, float* __restrict__ P)
{
    constexpr int Kd = 128;
    constexpr int KB = Kd / 32;          // 4 K-blocks of 32
    constexpr int Nd = NT * 16;
    __shared__ _Float16 Wl[Kd * Nd];     // swizzled weight tile (<= 32 KB)

    const int tid = threadIdx.x;
    // stage + swizzle: dst i = ((((nt*KB + kb)*2 + g)*16 + n)*16 + t)
    //                  src   = Wg[(kb*32 + 16g + t)*Nd + nt*16 + n]
    for (int i = tid; i < Kd * Nd; i += 256) {
        int t  = i & 15;
        int n  = (i >> 4) & 15;
        int g  = (i >> 8) & 1;
        int kb = (i >> 9) & (KB - 1);
        int nt = i >> 11;
        Wl[i] = Wg[(kb * 32 + 16 * g + t) * Nd + nt * 16 + n];
    }
    __syncthreads();

    const int wave = tid >> 5;
    const int lane = tid & 31;
    const int g    = lane >> 4;          // half-wave group
    const int l16  = lane & 15;
    const int m0   = (blockIdx.x * 8 + wave) * 16;

    v8f acc[NT];
#pragma unroll
    for (int nt = 0; nt < NT; ++nt) acc[nt] = (v8f){};

    const _Float16* Xrow = X + (size_t)(m0 + l16) * Kd;

#pragma unroll
    for (int kb = 0; kb < KB; ++kb) {
        // A fragment: lane row m0+l16; VGPR0..7 = K kb*32+16g..+15
        v16h a;
        *(v8h*)&a         = *(const v8h*)(Xrow + kb * 32 + 16 * g);
        *((v8h*)&a + 1)   = *(const v8h*)(Xrow + kb * 32 + 16 * g + 8);
#pragma unroll
        for (int nt = 0; nt < NT; ++nt) {
            const _Float16* bp = &Wl[(((nt * KB + kb) * 2 + g) * 16 + l16) * 16];
            v16h b;
            *(v8h*)&b       = *(const v8h*)(bp);
            *((v8h*)&b + 1) = *(const v8h*)(bp + 8);
            acc[nt] = __builtin_amdgcn_wmma_f32_16x16x32_f16(
                false, a, false, b, (short)0, acc[nt], false, false);
        }
    }

    // D layout: lanes 0-15 -> M = m0 + r, lanes 16-31 -> M = m0 + 8 + r; N = nt*16 + l16
#pragma unroll
    for (int r = 0; r < 8; ++r) {
        const int m = m0 + 8 * g + r;
        const float s = od[m];
        float* prow = P + (size_t)m * Nd + l16;
#pragma unroll
        for (int nt = 0; nt < NT; ++nt) prow[nt * 16] = acc[nt][r] * s;
    }
}

// ---------------- edge scatter-add: agg[dst[e], c] += P[src[e], c] ----------------
template<int D, int LOGD>
__global__ void k_aggregate(const int* __restrict__ src, const int* __restrict__ dst,
                            const float* __restrict__ P, float* __restrict__ agg,
                            long long tot) {
    long long i = (long long)blockIdx.x * blockDim.x + threadIdx.x;
    if (i >= tot) return;
    int e = (int)(i >> LOGD);
    int c = (int)(i & (D - 1));
    const float* prow = P + (size_t)src[e] * D;
    __builtin_prefetch(prow + c, 0, 1);           // global_prefetch_b8 hint
    atomicAdd(&agg[(size_t)dst[e] * D + c], prow[c]);
}

// ---------------- finalize: h = relu(agg * id + b); emit f16 (and f32 on last) ----
template<int D, int LOGD>
__global__ void k_finalize(const float* __restrict__ agg, const float* __restrict__ idg,
                           const float* __restrict__ bias, _Float16* __restrict__ Xh,
                           float* __restrict__ h32, int writeF32) {
    long long i = (long long)blockIdx.x * blockDim.x + threadIdx.x;
    constexpr long long tot = (long long)TOTAL_NODES * D;
    if (i >= tot) return;
    int node = (int)(i >> LOGD);
    int c    = (int)(i & (D - 1));
    float v = agg[i] * idg[node] + bias[c];
    v = fmaxf(v, 0.0f);
    Xh[i] = (_Float16)v;
    if (writeF32) h32[i] = v;
}

// ---------------- per-node ascending sort of 64 features (in place) ----------------
__global__ void k_sort64(float* __restrict__ h, float* __restrict__ keys) {
    int node = blockIdx.x * blockDim.x + threadIdx.x;
    if (node >= TOTAL_NODES) return;
    float a[64];
    float* row = h + (size_t)node * 64;
    for (int i = 0; i < 64; ++i) a[i] = row[i];
    for (int i = 1; i < 64; ++i) {
        float v = a[i];
        int j = i - 1;
        while (j >= 0 && a[j] > v) { a[j + 1] = a[j]; --j; }
        a[j + 1] = v;
    }
    for (int i = 0; i < 64; ++i) row[i] = a[i];
    keys[node] = a[63];
}

// ---------------- per-graph top-16 by last feature, then gather ----------------
__global__ __launch_bounds__(256) void k_topk_gather(
    const float* __restrict__ hs, const float* __restrict__ keys,
    float* __restrict__ out) {
    __shared__ float sv[256];
    __shared__ int   si[256];
    __shared__ int   sel[KSORT];
    __shared__ unsigned char taken[NODES_PER_G];

    const int g = blockIdx.x, tid = threadIdx.x;
    for (int i = tid; i < NODES_PER_G; i += 256) taken[i] = 0;
    __syncthreads();

    for (int it = 0; it < KSORT; ++it) {
        float bv = -INFINITY; int bi = NODES_PER_G;
        for (int i = tid; i < NODES_PER_G; i += 256) {
            if (!taken[i]) {
                float v = keys[g * NODES_PER_G + i];
                if (v > bv || (v == bv && i < bi)) { bv = v; bi = i; }
            }
        }
        sv[tid] = bv; si[tid] = bi;
        __syncthreads();
        for (int s = 128; s > 0; s >>= 1) {
            if (tid < s) {
                float ov = sv[tid + s]; int oi = si[tid + s];
                if (ov > sv[tid] || (ov == sv[tid] && oi < si[tid])) { sv[tid] = ov; si[tid] = oi; }
            }
            __syncthreads();
        }
        if (tid == 0) { sel[it] = si[0]; taken[si[0] & (NODES_PER_G - 1)] = 1; }
        __syncthreads();
    }
    for (int i = tid; i < KSORT * 64; i += 256) {
        int it = i >> 6;
        out[(size_t)g * (KSORT * 64) + i] =
            hs[((size_t)g * NODES_PER_G + sel[it]) * 64 + (i & 63)];
    }
}

// ---------------------------------------------------------------------------

static inline int cdiv(long long a, int b) { return (int)((a + b - 1) / b); }

extern "C" void kernel_launch(void* const* d_in, const int* in_sizes, int n_in,
                              void* d_out, int out_size, void* d_ws, size_t ws_size,
                              hipStream_t stream) {
    const float* feat = (const float*)d_in[0];
    const int*   src  = (const int*)d_in[1];
    const int*   dst  = (const int*)d_in[2];
    const float* W[4]   = { (const float*)d_in[3], (const float*)d_in[5],
                            (const float*)d_in[7], (const float*)d_in[9] };
    const float* bia[4] = { (const float*)d_in[4], (const float*)d_in[6],
                            (const float*)d_in[8], (const float*)d_in[10] };
    const int E = in_sizes[1];
    float* out = (float*)d_out;

    // workspace carve-up (all 256B aligned)
    char* w = (char*)d_ws;
    auto take = [&](size_t bytes) { char* p = w; w += (bytes + 255) & ~(size_t)255; return p; };
    float*    od    = (float*)take((size_t)TOTAL_NODES * 4);
    float*    idg   = (float*)take((size_t)TOTAL_NODES * 4);
    float*    keys  = (float*)take((size_t)TOTAL_NODES * 4);
    _Float16* Wh    = (_Float16*)take((size_t)DMAX * DMAX * 2);
    _Float16* Xh    = (_Float16*)take((size_t)TOTAL_NODES * DMAX * 2);
    float*    P     = (float*)take((size_t)TOTAL_NODES * DMAX * 4);
    float*    agg   = (float*)take((size_t)TOTAL_NODES * DMAX * 4);
    float*    hlast = (float*)take((size_t)TOTAL_NODES * 64 * 4);
    (void)ws_size;

    // 1) degrees + normalizers
    k_zero_f32<<<cdiv(TOTAL_NODES, 256), 256, 0, stream>>>(od, TOTAL_NODES);
    k_zero_f32<<<cdiv(TOTAL_NODES, 256), 256, 0, stream>>>(idg, TOTAL_NODES);
    k_degrees<<<cdiv(E, 256), 256, 0, stream>>>(src, dst, od, idg, E);
    k_rsqrt_deg<<<cdiv(TOTAL_NODES, 256), 256, 0, stream>>>(od, TOTAL_NODES);
    k_rsqrt_deg<<<cdiv(TOTAL_NODES, 256), 256, 0, stream>>>(idg, TOTAL_NODES);

    // 2) features -> f16
    k_f32_to_f16<<<cdiv((long long)TOTAL_NODES * DMAX, 256), 256, 0, stream>>>(
        feat, Xh, TOTAL_NODES * DMAX);

    // 3) four GraphConv layers (K = 128; Nd = 128,128,128,64)
    const int douts[4] = { 128, 128, 128, 64 };
    for (int l = 0; l < 4; ++l) {
        const int Nd = douts[l];
        k_f32_to_f16<<<cdiv(128 * Nd, 256), 256, 0, stream>>>(W[l], Wh, 128 * Nd);

        const int gemm_blocks = TOTAL_NODES / (16 * 8);   // 512
        if (Nd == 128)
            k_gemm_wmma<8><<<gemm_blocks, 256, 0, stream>>>(Xh, Wh, od, P);
        else
            k_gemm_wmma<4><<<gemm_blocks, 256, 0, stream>>>(Xh, Wh, od, P);

        k_zero_f32<<<cdiv((long long)TOTAL_NODES * Nd, 256), 256, 0, stream>>>(
            agg, TOTAL_NODES * Nd);

        const long long etot = (long long)E * Nd;
        if (Nd == 128)
            k_aggregate<128, 7><<<cdiv(etot, 256), 256, 0, stream>>>(src, dst, P, agg, etot);
        else
            k_aggregate<64, 6><<<cdiv(etot, 256), 256, 0, stream>>>(src, dst, P, agg, etot);

        const long long ftot = (long long)TOTAL_NODES * Nd;
        if (Nd == 128)
            k_finalize<128, 7><<<cdiv(ftot, 256), 256, 0, stream>>>(agg, idg, bia[l], Xh, hlast, 0);
        else
            k_finalize<64, 6><<<cdiv(ftot, 256), 256, 0, stream>>>(agg, idg, bia[l], Xh, hlast, 1);
    }

    // 4) sortpooling
    k_sort64<<<cdiv(TOTAL_NODES, 256), 256, 0, stream>>>(hlast, keys);
    k_topk_gather<<<GRAPHS, 256, 0, stream>>>(hlast, keys, out);
    (void)out_size; (void)n_in;
}
